// XAIGuiAttention_10325101380374
// MI455X (gfx1250) — matrix operand order repaired and checked by reading, not compiled, not run on hardware
//
#include <hip/hip_runtime.h>

// ---------------------------------------------------------------------------
// MI455X (gfx1250, wave32) fused attention pipeline.
// ~206 GFLOP vs ~200MB HBM traffic -> matrix-core bound: all matmuls run on
// v_wmma_f32_16x16x32_f16 (f32 accum). Flash-attention keeps the SxS scores
// on-chip. Softmax row reductions are DPP16 butterflies (VALU, not DS pipe).
// K/V tiles are fetched by the Tensor Data Mover (TENSOR_LOAD_TO_LDS) with
// double buffering: TDM DMAs tile i+1 into LDS (using the D# pad feature to
// produce 72-half padded rows) while the waves run WMMAs on tile i; the
// issuing wave synchronizes with s_wait_tensorcnt. Workspace (f16):
// Q | K | V ([B,H,S,64]) | ctx ([B,S,1024]) = 128MB.
// ---------------------------------------------------------------------------

typedef __attribute__((ext_vector_type(16))) _Float16 v16h;
typedef __attribute__((ext_vector_type(4)))  _Float16 v4h;
typedef __attribute__((ext_vector_type(8)))  float    v8f;
typedef __attribute__((ext_vector_type(4)))  float    v4f;
typedef __attribute__((ext_vector_type(4)))  unsigned int v4u;
typedef __attribute__((ext_vector_type(8)))  int      v8i;
typedef __attribute__((ext_vector_type(4)))  int      v4i;

#if defined(__has_builtin)
#if __has_builtin(__builtin_amdgcn_tensor_load_to_lds) && __has_builtin(__builtin_amdgcn_s_wait_tensorcnt)
#define USE_TDM 1
#endif
#endif
#ifndef USE_TDM
#define USE_TDM 0
#endif

__device__ __forceinline__ v8f wmma_f16(v16h a, v16h b, v8f c) {
  // 8 args: (neg_a, A, neg_b, B, c_mod, C, reuse_a, reuse_b)
  return __builtin_amdgcn_wmma_f32_16x16x32_f16(false, a, false, b, (short)0, c,
                                                false, false);
}

#if USE_TDM
// TDM 2-D f16 tile load: tile_d0 x tile_d1 elements from a row-major tensor
// (row stride = stride0 elements) into LDS at lds_addr. D# pad feature adds
// 4 DWORDs after every 32 DWORDs stored -> rows land on a 72-half pitch.
// D# layout per CDNA5 ISA 8.3/8.4 (group0 128b, group1 256b; groups 2/3 zero
// for 2-D tiles).
__device__ __forceinline__ void tdm_load_tile_f16(
    unsigned lds_addr, const _Float16* gptr,
    int tile_d0, int tile_d1, int tensor_d0, int tensor_d1, int stride0) {
  const unsigned long long ga = (unsigned long long)(uintptr_t)gptr;
  v4u g0;
  g0.x = 1u;                                   // count=1, user descriptor
  g0.y = lds_addr;                             // LDS byte address
  g0.z = (unsigned)(ga & 0xFFFFFFFFu);         // global_addr[31:0]
  g0.w = (unsigned)((ga >> 32) & 0x01FFFFFFu)  // global_addr[56:32]
       | (2u << 30);                           // type=2 ("image")
  v8i g1;
  g1[0] = (1 << 16)                            // data_size = 2 bytes
        | (1 << 20)                            // pad_enable
        | (4 << 22)                            // pad_interval: 32 DWORDs
        | (3 << 25);                           // pad_amount: 4 DWORDs (8 halfs)
  g1[1] = (tensor_d0 & 0xFFFF) << 16;          // tensor_dim0[15:0] @ bits 63:48
  g1[2] = ((tensor_d0 >> 16) & 0xFFFF) | ((tensor_d1 & 0xFFFF) << 16);
  g1[3] = ((tensor_d1 >> 16) & 0xFFFF) | ((tile_d0 & 0xFFFF) << 16);
  g1[4] = (tile_d1 & 0xFFFF);                  // tile_dim1; tile_dim2 = 0
  g1[5] = stride0;                             // tensor_dim0_stride[31:0]
  g1[6] = 0;
  g1[7] = 0;
  const v4i z4 = {0, 0, 0, 0};
#if __clang_major__ >= 23
  const v8i z8 = {0, 0, 0, 0, 0, 0, 0, 0};
  __builtin_amdgcn_tensor_load_to_lds(g0, g1, z4, z4, z8, 0);
#else
  __builtin_amdgcn_tensor_load_to_lds(g0, g1, z4, z4, 0);
#endif
}
__device__ __forceinline__ unsigned lds_off(const void* p) {
  return (unsigned)(uintptr_t)p;  // low 32 bits of a generic LDS addr = offset
}
#endif  // USE_TDM

// ---- DPP16 16-lane butterfly reductions (lanes 0-15 / 16-31 reduce
// independently, matching C-fragment n-striping). quad_perm[1,0,3,2]=0xB1,
// quad_perm[2,3,0,1]=0x4E, row_half_mirror=0x141, row_mirror=0x140.
template <int CTRL>
__device__ __forceinline__ float dpp_f(float x) {
  const int i = __float_as_int(x);
  return __int_as_float(__builtin_amdgcn_update_dpp(i, i, CTRL, 0xF, 0xF, true));
}
__device__ __forceinline__ float red_max16(float x) {
  x = fmaxf(x, dpp_f<0xB1>(x));
  x = fmaxf(x, dpp_f<0x4E>(x));
  x = fmaxf(x, dpp_f<0x141>(x));
  x = fmaxf(x, dpp_f<0x140>(x));
  return x;
}
__device__ __forceinline__ float red_sum16(float x) {
  x += dpp_f<0xB1>(x);
  x += dpp_f<0x4E>(x);
  x += dpp_f<0x141>(x);
  x += dpp_f<0x140>(x);
  return x;
}

// ---- vectorized staging helpers
__device__ __forceinline__ v4f load4(const float* p) { return *(const v4f*)p; }
__device__ __forceinline__ v4f load4(const _Float16* p) {
  const v4h t = *(const v4h*)p;
  return (v4f){(float)t.x, (float)t.y, (float)t.z, (float)t.w};
}
__device__ __forceinline__ v4h to_h4(v4f v) {
  return (v4h){(_Float16)v.x, (_Float16)v.y, (_Float16)v.z, (_Float16)v.w};
}

// ---- WMMA fragment gathers from LDS (contiguous k = (lane>>4)*16 + j, same in A and B) ----
__device__ __forceinline__ v16h frag_a(const _Float16* lds, int ld, int k0) {
  const int lane = threadIdx.x & 31;
  const int m = lane & 15;
  const int hf = lane >> 4;
  v16h a;
#pragma unroll
  for (int j = 0; j < 16; ++j) {
    const int k = hf * 16 + j;
    a[j] = lds[m * ld + k0 + k];
  }
  return a;
}
// B frag from N-major tile (tile[n][k]); element j: K = (lane>>4)*16+j -> b128.
__device__ __forceinline__ v16h frag_b_nmajor(const _Float16* lds, int ld, int n0, int k0) {
  const int lane = threadIdx.x & 31;
  const int n = n0 + (lane & 15);
  const int hf = lane >> 4;
  v16h b;
#pragma unroll
  for (int j = 0; j < 16; ++j) b[j] = lds[n * ld + k0 + hf * 16 + j];
  return b;
}
// B frag from K-major tile (tile[k][n]).
__device__ __forceinline__ v16h frag_b_kmajor(const _Float16* lds, int ld, int n0, int k0) {
  const int lane = threadIdx.x & 31;
  const int n = n0 + (lane & 15);
  const int hf = lane >> 4;
  v16h b;
#pragma unroll
  for (int j = 0; j < 16; ++j) b[j] = lds[(k0 + hf * 16 + j) * ld + n];
  return b;
}

// ---------------------------------------------------------------------------
// C = X * W^T + bias.  X:[M,K] (TIN), W:[N,K] fp32, bias:[N] fp32.
// 256 threads (8 wave32), C-tile 128x128, k-step 32. (fp32->f16 conversion
// happens on load, so staging stays on the VALU path, not TDM.)
// ---------------------------------------------------------------------------
template <int HEAD_SPLIT, typename TIN, typename TOUT>
__global__ void __launch_bounds__(256) gemm_xwt_bias(
    const TIN* __restrict__ X, const float* __restrict__ W,
    const float* __restrict__ bias, TOUT* __restrict__ out,
    int M, int N, int K) {
  __shared__ _Float16 Xs[128][40];
  __shared__ _Float16 Ws[128][40];
  const int tid = threadIdx.x;
  const int wv = tid >> 5;
  const int lane = tid & 31;
  const int m0 = blockIdx.x * 128;
  const int n0 = blockIdx.y * 128;

  v8f acc[8] = {};

  for (int k0 = 0; k0 < K; k0 += 32) {
    __syncthreads();
#pragma unroll
    for (int i = 0; i < 4; ++i) {
      const int idx = (tid + i * 256) * 4;
      const int r = idx >> 5, c = idx & 31;
      *(v4h*)&Xs[r][c] = to_h4(load4(X + (size_t)(m0 + r) * K + k0 + c));
      *(v4h*)&Ws[r][c] = to_h4(load4(W + (size_t)(n0 + r) * K + k0 + c));
    }
    if (k0 + 32 < K) {  // -> global_prefetch_b8
      const int idx = tid * 4;
      const int r = idx >> 5, c = idx & 31;
      __builtin_prefetch(X + (size_t)(m0 + r) * K + k0 + 32 + c, 0, 0);
      __builtin_prefetch(W + (size_t)(n0 + r) * K + k0 + 32 + c, 0, 0);
    }
    __syncthreads();
    const v16h a = frag_a(&Xs[wv * 16][0], 40, 0);
#pragma unroll
    for (int nt = 0; nt < 8; ++nt) {
      const v16h b = frag_b_nmajor(&Ws[0][0], 40, nt * 16, 0);
      acc[nt] = wmma_f16(a, b, acc[nt]);
    }
  }

  const int nl = lane & 15;
  const int hf = lane >> 4;
#pragma unroll
  for (int nt = 0; nt < 8; ++nt) {
    const int gn = n0 + nt * 16 + nl;
    const float bval = bias[gn];
#pragma unroll
    for (int r = 0; r < 8; ++r) {
      const int gm = m0 + wv * 16 + r + 8 * hf;
      const float val = acc[nt][r] + bval;
      if (HEAD_SPLIT) {
        const int bb = gm >> 10, s = gm & 1023;   // S = 1024
        const int h = gn >> 6, hd = gn & 63;      // HD = 64
        out[((((size_t)bb * 16 + h) * 1024) + s) * 64 + hd] = (TOUT)val;
      } else {
        out[(size_t)gm * N + gn] = (TOUT)val;
      }
    }
  }
}

// ---------------------------------------------------------------------------
// Flash attention: one (b, h, 128-query-row) tile per 256-thread block.
// 64-key stages; K/V tiles double-buffered via TDM (tile i+1 in flight while
// computing on tile i). ctx (f16) written merged-head [B,S,1024].
// ---------------------------------------------------------------------------
__global__ void __launch_bounds__(256) attn_fwd(
    const _Float16* __restrict__ Q, const _Float16* __restrict__ K,
    const _Float16* __restrict__ V, _Float16* __restrict__ ctx, float scale) {
  const int tid = threadIdx.x;
  const int wv = tid >> 5;
  const int lane = tid & 31;
  const int nl = lane & 15;
  const int hf = lane >> 4;
  const int qb = blockIdx.x;
  const int h = blockIdx.y;
  const int b = blockIdx.z;
  const size_t bh = (size_t)b * 16 + h;
  const _Float16* Qp = Q + bh * 1024 * 64;
  const _Float16* Kp = K + bh * 1024 * 64;
  const _Float16* Vp = V + bh * 1024 * 64;

  __shared__ _Float16 Qs[128][72];
#if USE_TDM
  __shared__ _Float16 Ks[2][64][72];
  __shared__ _Float16 Vs[2][64][72];
#else
  __shared__ _Float16 Ks[1][64][72];
  __shared__ _Float16 Vs[1][64][72];
#endif
  __shared__ _Float16 Ps[8][16][72];  // per-wave P (C-layout -> A-layout bounce)

#if USE_TDM
  if (wv == 0) {  // TDM: Q tile (128x64) + first K/V tiles (64x64)
    tdm_load_tile_f16(lds_off(&Qs[0][0]), Qp + (size_t)qb * 128 * 64,
                      64, 128, 64, 1024, 64);
    tdm_load_tile_f16(lds_off(&Ks[0][0][0]), Kp, 64, 64, 64, 1024, 64);
    tdm_load_tile_f16(lds_off(&Vs[0][0][0]), Vp, 64, 64, 64, 1024, 64);
    __builtin_amdgcn_s_wait_tensorcnt((short)0);
  }
  __syncthreads();
#else
#pragma unroll
  for (int i = 0; i < 8; ++i) {
    const int idx = (tid + i * 256) * 4;
    const int r = idx >> 6, c = idx & 63;
    *(v4h*)&Qs[r][c] = *(const v4h*)(Qp + (size_t)(qb * 128 + r) * 64 + c);
  }
  __syncthreads();
#endif

  const v16h aQ0 = frag_a(&Qs[wv * 16][0], 72, 0);
  const v16h aQ1 = frag_a(&Qs[wv * 16][0], 72, 32);

  v8f acc[4] = {};
  float rmax[8], rsum[8];
#pragma unroll
  for (int r = 0; r < 8; ++r) { rmax[r] = -3.0e38f; rsum[r] = 0.0f; }

  for (int it = 0; it < 16; ++it) {
#if USE_TDM
    const int cur = it & 1;
    if (wv == 0) {
      if (it + 1 < 16) {  // prefetch next tile into the other buffer
        const size_t koff = (size_t)(it + 1) * 64 * 64;
        tdm_load_tile_f16(lds_off(&Ks[cur ^ 1][0][0]), Kp + koff, 64, 64, 64, 1024, 64);
        tdm_load_tile_f16(lds_off(&Vs[cur ^ 1][0][0]), Vp + koff, 64, 64, 64, 1024, 64);
        __builtin_amdgcn_s_wait_tensorcnt((short)2);  // current tile complete
      } else {
        __builtin_amdgcn_s_wait_tensorcnt((short)0);
      }
    }
    __syncthreads();  // current tile visible to all waves
    const _Float16* Kt = &Ks[cur][0][0];
    const _Float16* Vt = &Vs[cur][0][0];
#else
    __syncthreads();
#pragma unroll
    for (int i = 0; i < 4; ++i) {
      const int idx = (tid + i * 256) * 4;
      const int r = idx >> 6, c = idx & 63;
      *(v4h*)&Ks[0][r][c] = *(const v4h*)(Kp + (size_t)(it * 64 + r) * 64 + c);
      *(v4h*)&Vs[0][r][c] = *(const v4h*)(Vp + (size_t)(it * 64 + r) * 64 + c);
    }
    __syncthreads();
    const _Float16* Kt = &Ks[0][0][0];
    const _Float16* Vt = &Vs[0][0][0];
#endif

    // scores S = Q * K^T : K tile [key][hd] is N-major for this product.
    v8f s[4];
#pragma unroll
    for (int nt = 0; nt < 4; ++nt) {
      v8f z = {};
      z = wmma_f16(aQ0, frag_b_nmajor(Kt, 72, nt * 16, 0), z);
      z = wmma_f16(aQ1, frag_b_nmajor(Kt, 72, nt * 16, 32), z);
      s[nt] = z;
    }

    // online softmax (VGPR r -> row r + 8*hf; 16-lane n-groups)
    float mx[8];
#pragma unroll
    for (int r = 0; r < 8; ++r) {
      s[0][r] *= scale; s[1][r] *= scale; s[2][r] *= scale; s[3][r] *= scale;
      mx[r] = fmaxf(fmaxf(s[0][r], s[1][r]), fmaxf(s[2][r], s[3][r]));
      mx[r] = red_max16(mx[r]);
    }
    float psc[8];
#pragma unroll
    for (int r = 0; r < 8; ++r) {
      const float nm = fmaxf(rmax[r], mx[r]);
      psc[r] = __expf(rmax[r] - nm);
      rmax[r] = nm;
#pragma unroll
      for (int nt = 0; nt < 4; ++nt) s[nt][r] = __expf(s[nt][r] - nm);
    }
#pragma unroll
    for (int r = 0; r < 8; ++r) {
      const float rs = red_sum16((s[0][r] + s[1][r]) + (s[2][r] + s[3][r]));
      rsum[r] = rsum[r] * psc[r] + rs;
#pragma unroll
      for (int nt = 0; nt < 4; ++nt) acc[nt][r] *= psc[r];
    }

    // P (C-layout) -> per-wave LDS -> A-layout fragments
#pragma unroll
    for (int nt = 0; nt < 4; ++nt)
#pragma unroll
      for (int r = 0; r < 8; ++r)
        Ps[wv][r + 8 * hf][nt * 16 + nl] = (_Float16)s[nt][r];
    asm volatile("s_wait_dscnt 0" ::: "memory");  // same-wave LDS RAW fence
    const v16h aP0 = frag_a(&Ps[wv][0][0], 72, 0);
    const v16h aP1 = frag_a(&Ps[wv][0][0], 72, 32);

    // ctx += P * V : V tile [key][hd] is K-major for this product.
#pragma unroll
    for (int nt = 0; nt < 4; ++nt) {
      acc[nt] = wmma_f16(aP0, frag_b_kmajor(Vt, 72, nt * 16, 0), acc[nt]);
      acc[nt] = wmma_f16(aP1, frag_b_kmajor(Vt, 72, nt * 16, 32), acc[nt]);
    }
#if USE_TDM
    __syncthreads();  // buffer consumed; safe for TDM to refill next iteration
#endif
  }

  // normalize, store ctx in merged-head [B, S, 1024] layout
  _Float16* cp = ctx + ((size_t)b * 1024 + qb * 128 + wv * 16) * 1024 + h * 64;
#pragma unroll
  for (int r = 0; r < 8; ++r) {
    const float inv = 1.0f / rsum[r];
    const size_t ro = (size_t)(r + 8 * hf) * 1024;
#pragma unroll
    for (int nt = 0; nt < 4; ++nt)
      cp[ro + nt * 16 + nl] = (_Float16)(acc[nt][r] * inv);
  }
}

// ---------------------------------------------------------------------------
extern "C" void kernel_launch(void* const* d_in, const int* in_sizes, int n_in,
                              void* d_out, int out_size, void* d_ws, size_t ws_size,
                              hipStream_t stream) {
  const float* x  = (const float*)d_in[0];
  const float* ex = (const float*)d_in[1];
  const float* Wq = (const float*)d_in[2];
  const float* bq = (const float*)d_in[3];
  const float* Wk = (const float*)d_in[4];
  const float* bk = (const float*)d_in[5];
  const float* Wv = (const float*)d_in[6];
  const float* bv = (const float*)d_in[7];
  const float* Wo = (const float*)d_in[8];
  const float* bo = (const float*)d_in[9];
  float* out = (float*)d_out;

  const int Bn = 16, S = 1024, D = 1024, H = 16;
  const int M = Bn * S;                     // 16384
  const size_t elems = (size_t)Bn * S * D;  // 16M
  // d_ws: Q | K | V | ctx, each elems f16 (32 MB) -> 128 MB total.
  _Float16* qws = (_Float16*)d_ws;
  _Float16* kws = qws + elems;
  _Float16* vws = kws + elems;
  _Float16* cws = vws + elems;

  dim3 tb(256);
  dim3 gg(M / 128, D / 128);
  gemm_xwt_bias<1, float, _Float16><<<gg, tb, 0, stream>>>(ex, Wq, bq, qws, M, D, D);
  gemm_xwt_bias<1, float, _Float16><<<gg, tb, 0, stream>>>(ex, Wk, bk, kws, M, D, D);
  gemm_xwt_bias<1, float, _Float16><<<gg, tb, 0, stream>>>(x,  Wv, bv, vws, M, D, D);

  dim3 ga(S / 128, H, Bn);
  attn_fwd<<<ga, tb, 0, stream>>>(qws, kws, vws, cws, 0.125f /* 1/sqrt(64) */);

  gemm_xwt_bias<0, _Float16, float><<<gg, tb, 0, stream>>>(cws, Wo, bo, out, M, D, D);
}
